// MultiLayerLSTM_13288628814246
// MI455X (gfx1250) — matrix operand order repaired and checked by reading, not compiled, not run on hardware
//
#include <hip/hip_runtime.h>
#include <hip/hip_bf16.h>
#include <math.h>

// ---------------------------------------------------------------------------
// Problem constants (from reference): B=32, T=512, H=1024, L=4
// ---------------------------------------------------------------------------
#define BB   32
#define TT   512
#define HH   1024
#define LL   4
#define NWG  64      // workgroups: each owns 16 h-columns (64*16 = 1024 = H)
#define NTHR 1024    // 32 waves: (gate 0..3) x (mtile 0..1) x (K-quarter 0..3)

typedef __attribute__((ext_vector_type(16))) __bf16 v16bf;
typedef __attribute__((ext_vector_type(8)))  float  v8f;

// ---------------- workspace layout (bytes) ----------------
// A-fragment matrix (one [32 x 1024] activation set, bf16, ISA A-layout):
//   [chunk c(32)][mtile(2)][lane(32)][e(16)]  -> 32768 elems = 64 KB
#define MAT_ELEMS 32768
#define BAR_OFF   0
#define HBF_OFF   256
#define HBF_BYTES (2 * LL * MAT_ELEMS * 2)      // h fragments, dbl-buffered: 512 KB
#define ZERO_BYTES (HBF_OFF + HBF_BYTES)        // memset-0 region (bf16 0 == 0x0000)
#define BIAS_OFF  ZERO_BYTES
#define BIAS_BYTES (LL * 4 * HH * 4)            // combined bias: 64 KB
#define XBF_OFF   (BIAS_OFF + BIAS_BYTES)
#define XBF_BYTES (TT * MAT_ELEMS * 2)          // x fragments: 32 MB
#define WSWZ_OFF  (XBF_OFF + XBF_BYTES)         // bf16 swizzled weights: 64 MB
// wswz flat index: ((((l*64+wg)*4+g)*64 + chunk)*32 + lane)*16 + e
//   chunk 0..63 over concat-K [W_ih (k<1024) | W_hh (k>=1024)]
//   B-matrix ISA layout: n = g*1024 + wg*16 + (lane&15),
//   k_in_matrix = (chunk&31)*32 + 16*(e>>3) + 8*(lane>>4) + (e&7)

__device__ __forceinline__ float sigmoidf_(float x) {
    return 1.0f / (1.0f + __expf(-x));
}
// branch-free tanh: t = (1-e)/(1+e), e = exp(-2|x|); sign restored
__device__ __forceinline__ float tanhf_(float x) {
    float e = __expf(-2.0f * fabsf(x));
    float t = (1.0f - e) / (1.0f + e);
    return copysignf(t, x);
}

// ---------------------------------------------------------------------------
// Prep kernel 1: combined bias = b_ih + b_hh   (L*4H elements)
// ---------------------------------------------------------------------------
__global__ void lstm_bias_prep(const float* __restrict__ b_ih,
                               const float* __restrict__ b_hh,
                               float* __restrict__ bias) {
    int i = blockIdx.x * 256 + threadIdx.x;   // 16384 total
    bias[i] = b_ih[i] + b_hh[i];
}

// ---------------------------------------------------------------------------
// Prep kernel 2: fp32 weights -> bf16 in WMMA B-matrix lane layout.
// One thread emits one lane's 16 contiguous bf16.  2,097,152 lane-vectors.
// ---------------------------------------------------------------------------
__global__ void lstm_wswz_prep(const float* __restrict__ W_ih,
                               const float* __restrict__ W_hh,
                               __bf16* __restrict__ wswz) {
    size_t v = (size_t)blockIdx.x * 256 + threadIdx.x;
    int lane  = (int)(v & 31);
    int chunk = (int)((v >> 5) & 63);
    int g     = (int)((v >> 11) & 3);
    int wg    = (int)((v >> 13) & 63);
    int l     = (int)(v >> 19);
    int kh    = chunk >> 5;                       // 0 = W_ih, 1 = W_hh
    const float* src = (kh ? W_hh : W_ih) + (size_t)l * (4 * HH) * HH;
    int n    = g * HH + wg * 16 + (lane & 15);
    int kmat = (chunk & 31) * 32 + ((lane >> 4) << 3);
    const float* s = src + (size_t)n * HH + kmat;
    __bf16* d = wswz + v * 16;
#pragma unroll
    for (int e = 0; e < 16; ++e) d[e] = (__bf16)s[((e >> 3) << 4) + (e & 7)];
}

// ---------------------------------------------------------------------------
// Prep kernel 3: x_seq fp32 -> bf16 A-fragment layout, per timestep.
// xbf[t][c][mt][lane][e];  k_in_chunk = 16*(e>>3) + 8*(lane>>4) + (e&7)
// 1,048,576 lane-vectors.
// ---------------------------------------------------------------------------
__global__ void lstm_xbf_prep(const float* __restrict__ xseq,
                              __bf16* __restrict__ xbf) {
    size_t v = (size_t)blockIdx.x * 256 + threadIdx.x;
    int lane = (int)(v & 31);
    int mt   = (int)((v >> 5) & 1);
    int c    = (int)((v >> 6) & 31);
    int t    = (int)(v >> 11);
    int m    = mt * 16 + (lane & 15);
    const float* s = xseq + ((size_t)m * TT + t) * HH + c * 32 + ((lane >> 4) << 3);
    __bf16* d = xbf + v * 16;
#pragma unroll
    for (int e = 0; e < 16; ++e) d[e] = (__bf16)s[((e >> 3) << 4) + (e & 7)];
}

// ---------------------------------------------------------------------------
// Device-wide barrier (generation counter).  64 blocks co-resident.
// ---------------------------------------------------------------------------
__device__ __forceinline__ void grid_barrier(unsigned* bar, unsigned nblocks) {
    __syncthreads();
    if (threadIdx.x == 0) {
        unsigned gen = __hip_atomic_load(&bar[1], __ATOMIC_RELAXED,
                                         __HIP_MEMORY_SCOPE_AGENT);
        __threadfence();
        unsigned arrived = __hip_atomic_fetch_add(&bar[0], 1u, __ATOMIC_ACQ_REL,
                                                  __HIP_MEMORY_SCOPE_AGENT);
        if (arrived == nblocks - 1) {
            __hip_atomic_store(&bar[0], 0u, __ATOMIC_RELAXED,
                               __HIP_MEMORY_SCOPE_AGENT);
            __hip_atomic_fetch_add(&bar[1], 1u, __ATOMIC_RELEASE,
                                   __HIP_MEMORY_SCOPE_AGENT);
        } else {
            while (__hip_atomic_load(&bar[1], __ATOMIC_RELAXED,
                                     __HIP_MEMORY_SCOPE_AGENT) == gen) {
                __builtin_amdgcn_s_sleep(4);
            }
        }
        __threadfence();
    }
    __syncthreads();
}

// ---------------------------------------------------------------------------
// Persistent LSTM kernel.  64 blocks x 1024 threads.
// Block wg owns h-columns [wg*16, wg*16+16).  Wave w: gate g = w&3,
// mtile = (w>>2)&1, K-quarter kq = w>>3.  All operands are pre-laid-out
// WMMA fragments in global (L2-resident) -> K loop has NO barriers/LDS.
// ---------------------------------------------------------------------------
__global__ __launch_bounds__(NTHR, 1)
void lstm_persistent(const float* __restrict__ bias,
                     const __bf16* __restrict__ xbf,    // [T][frag]
                     const __bf16* __restrict__ wswz,
                     __bf16* __restrict__ hbf,          // [2][L][frag]
                     float* __restrict__ out,           // y | h_f | c_f
                     unsigned* __restrict__ bar) {
    __shared__ float d_part[32 * 256];                  // [wave][lane*8+v]

    const int tid  = threadIdx.x;
    const int wg   = blockIdx.x;
    const int wave = tid >> 5;
    const int lane = tid & 31;
    const int g    = wave & 3;
    const int mt   = (wave >> 2) & 1;
    const int kq   = wave >> 3;

    // ---- pointwise thread-private state ----
    const int pm   = tid >> 4;          // batch row (valid for tid<512)
    const int pn16 = tid & 15;
    const int pcol = wg * 16 + pn16;
    float creg[LL];                      // c state lives in registers
    float breg[LL][4];                   // biases preloaded
#pragma unroll
    for (int l = 0; l < LL; ++l) {
        creg[l] = 0.0f;
#pragma unroll
        for (int g2 = 0; g2 < 4; ++g2)
            breg[l][g2] = bias[((size_t)l * 4 + g2) * HH + pcol];
    }
    // h-fragment store position for (pm, pcol)
    const int pkk    = pcol & 31;
    const int pc     = pcol >> 5;
    const int plane  = (pm & 15) | (((pkk >> 3) & 1) << 4);
    const int pe     = ((pkk >> 4) << 3) | (pkk & 7);
    const size_t pfi = (((size_t)pc * 2 + ((pm >> 4) & 1)) * 32 + plane) * 16 + pe;

    const v16bf* wsv = (const v16bf*)wswz;
    const v16bf* hbv = (const v16bf*)hbf;
    const v16bf* xbv = (const v16bf*)xbf;

    for (int t = 0; t < TT; ++t) {
        const int p  = t & 1;
        const int p1 = p ^ 1;
#pragma unroll 1   // keep layers as a runtime loop: stops LICM from hoisting
        for (int l = 0; l < LL; ++l) {     // all weight fragments -> no spills
            // ---- A-fragment base for this wave's K-quarter ----
            const v16bf* av;
            int cstart;
            if (kq < 2) {      // x-part of concat K
                av = (l == 0) ? xbv + (size_t)t * (MAT_ELEMS / 16)
                              : hbv + (size_t)(p1 * LL + (l - 1)) * (MAT_ELEMS / 16);
                cstart = kq * 16;
            } else {           // h-part (previous step's h[l])
                av = hbv + (size_t)(p * LL + l) * (MAT_ELEMS / 16);
                cstart = (kq - 2) * 16;
            }
            av += (size_t)(cstart * 2 + mt) * 32 + lane;

            const v16bf* wp = wsv +
                ((((size_t)l * NWG + wg) * 4 + g) * 64 + (size_t)kq * 16) * 32 + lane;

            v8f acc = {};
#pragma unroll 4   // 4 A/B pairs in flight (64 VGPRs) -- no register spills
            for (int kci = 0; kci < 16; ++kci) {
                v16bf a = av[kci * 64];     // 2x global_load_b128 (A frag)
                v16bf b = wp[kci * 32];     // 2x global_load_b128 (B frag)
                acc = __builtin_amdgcn_wmma_f32_16x16x32_bf16(
                    false, a, false, b, (short)0, acc, false, false);
            }

            // ---- spill partial tile (ISA D layout, 32B/lane) ----
            *(v8f*)(d_part + wave * 256 + lane * 8) = acc;
            __syncthreads();

            // ---- pointwise LSTM cell on 512 threads ----
            if (tid < 512) {
                int mtp = pm >> 4;
                int v   = pm & 7;
                int lh  = (pm >> 3) & 1;
                int ln  = pn16 + (lh << 4);

                float gv[4];
#pragma unroll
                for (int g2 = 0; g2 < 4; ++g2) {
                    float s = breg[l][g2];
                    int wb = mtp * 4 + g2;
#pragma unroll
                    for (int q2 = 0; q2 < 4; ++q2)
                        s += d_part[((q2 << 3) + wb) * 256 + ln * 8 + v];
                    gv[g2] = s;
                }
                float ig = sigmoidf_(gv[0]);
                float fg = sigmoidf_(gv[1]);
                float gg = tanhf_(gv[2]);
                float og = sigmoidf_(gv[3]);

                float cnew = fg * creg[l] + ig * gg;
                float hnew = og * tanhf_(cnew);
                creg[l] = cnew;

                // h_new -> bf16 A-fragment (next parity) for the GEMM consumers
                hbf[(size_t)(p1 * LL + l) * MAT_ELEMS + pfi] = (__bf16)hnew;

                if (l == LL - 1)
                    out[((size_t)pm * TT + t) * HH + pcol] = hnew;
                if (t == TT - 1) {
                    size_t cidx = ((size_t)l * BB + pm) * HH + pcol;
                    out[(size_t)BB * TT * HH + cidx] = hnew;                        // h_f
                    out[(size_t)BB * TT * HH + (size_t)LL * BB * HH + cidx] = cnew; // c_f
                }
            }
            grid_barrier(bar, gridDim.x);
        }
    }
}

// ---------------------------------------------------------------------------
extern "C" void kernel_launch(void* const* d_in, const int* in_sizes, int n_in,
                              void* d_out, int out_size, void* d_ws, size_t ws_size,
                              hipStream_t stream) {
    const float* xseq = (const float*)d_in[0];
    const float* W_ih = (const float*)d_in[1];
    const float* W_hh = (const float*)d_in[2];
    const float* b_ih = (const float*)d_in[3];
    const float* b_hh = (const float*)d_in[4];

    char* ws = (char*)d_ws;
    unsigned* bar = (unsigned*)(ws + BAR_OFF);
    __bf16* hbf  = (__bf16*)(ws + HBF_OFF);
    float*  bias = (float*)(ws + BIAS_OFF);
    __bf16* xbf  = (__bf16*)(ws + XBF_OFF);
    __bf16* wswz = (__bf16*)(ws + WSWZ_OFF);
    float*  out  = (float*)d_out;

    // zero barrier + h fragment state (deterministic every call; capture-safe)
    hipMemsetAsync(ws, 0, ZERO_BYTES, stream);

    // re-derive fragments every call (no caching, graph-capture safe)
    lstm_bias_prep<<<(LL * 4 * HH) / 256, 256, 0, stream>>>(b_ih, b_hh, bias);
    lstm_wswz_prep<<<8192, 256, 0, stream>>>(W_ih, W_hh, wswz);
    lstm_xbf_prep<<<4096, 256, 0, stream>>>(xseq, xbf);

    // persistent LSTM: 64 co-resident blocks, device-wide barrier per phase
    lstm_persistent<<<NWG, NTHR, 0, stream>>>(bias, xbf, wswz, hbf, out, bar);
}
